// GNN_14946486190734
// MI455X (gfx1250) — matrix operand order repaired and checked by reading, not compiled, not run on hardware
//
#include <hip/hip_runtime.h>

// ---------------------------------------------------------------------------
// Fused 2-layer SAGE(pool) chain + edge products for gfx1250 (MI455X).
// All GEMMs are [rows x 64] * [64 x 64] -> bf16 WMMA (v_wmma_f32_16x16x32_bf16)
// with f32 accumulate. One workgroup = 256 threads = 8 wave32 = 8 16-row
// M-tiles = 124 valid output rows (+/-2 halo). Entire chain staged in LDS
// (~209 KB of the 320 KB/WGP).
// ---------------------------------------------------------------------------

typedef __attribute__((ext_vector_type(16))) __bf16 v16bf;
typedef __attribute__((ext_vector_type(8)))  __bf16 v8bf;
typedef __attribute__((ext_vector_type(8)))  float  v8f;

#define BSZ     64
#define LSZ     4096
#define R_ROWS  128          // rows staged per workgroup (8 tiles of 16)
#define T_OUT   124          // valid output rows per workgroup (halo = 2+2)
#define BF_S    80           // bf16 LDS row stride (160B, 16B aligned, bank-spread)
#define F_S     68           // f32 LDS row stride (272B, bank-spread)
#define W_S     80           // transposed-weight bf16 row stride

static constexpr float DIST_MEAN = 0.274716042312f;
static constexpr float DIST_STD  = 0.127051674693f;

static_assert(R_ROWS == 8 * 16, "8 waves x 16-row tiles");

// A fragment (bf16 16x32): lane m+16h holds row m; elems 0..7 <- K=koff+8h..,
// elems 8..15 <- K=koff+16+8h.. (ISA 7.12.2, 16-bit A 16x32 table).
__device__ __forceinline__ v16bf load_afrag(const __bf16* row, int h, int koff) {
    union { v16bf v; v8bf p[2]; } u;
    u.p[0] = *(const v8bf*)(row + koff + 8 * h);
    u.p[1] = *(const v8bf*)(row + koff + 16 + 8 * h);
    return u.v;
}

// B fragment (bf16 32x16): lane n+16h holds column n; elems 0..7 <- K=koff+8h..,
// elems 8..15 <- K=koff+16+8h.. -> weights stored transposed ([N][K]), two 16B runs.
__device__ __forceinline__ v16bf load_bfrag(const __bf16* wrow, int h, int koff) {
    union { v16bf v; v8bf p[2]; } u;
    u.p[0] = *(const v8bf*)(wrow + koff + 8 * h);
    u.p[1] = *(const v8bf*)(wrow + koff + 16 + 8 * h);
    return u.v;
}

// p = relu(src @ W + b), written as bf16; rows with global index < 0 zeroed
// (DGL zero-fill for the shift).  8 WMMA per wave.
__device__ __forceinline__ void stage_pool(const __bf16* __restrict__ src,
                                           const __bf16* __restrict__ Wt,
                                           const float*  __restrict__ bias,
                                           __bf16* __restrict__ dst,
                                           int r0, int wave, int n, int h) {
    const __bf16* ar = src + (wave * 16 + n) * BF_S;   // lane&15 doubles as M row
    v16bf a0 = load_afrag(ar, h, 0);
    v16bf a1 = load_afrag(ar, h, 32);
#pragma unroll
    for (int nt = 0; nt < 4; ++nt) {
        const __bf16* wr = Wt + (nt * 16 + n) * W_S;
        float bv = bias[nt * 16 + n];
        v8f acc;
#pragma unroll
        for (int i = 0; i < 8; ++i) acc[i] = bv;
        v16bf b0 = load_bfrag(wr, h, 0);
        v16bf b1 = load_bfrag(wr, h, 32);
        acc = __builtin_amdgcn_wmma_f32_16x16x32_bf16(false, a0, false, b0, (short)0, acc, false, false);
        acc = __builtin_amdgcn_wmma_f32_16x16x32_bf16(false, a1, false, b1, (short)0, acc, false, false);
#pragma unroll
        for (int r = 0; r < 8; ++r) {               // D: VGPR r -> row r+8h, col n
            int  orow = wave * 16 + r + 8 * h;
            long g    = (long)r0 - 2 + orow;
            float v   = fmaxf(acc[r], 0.0f);
            dst[orow * BF_S + nt * 16 + n] = (g < 0) ? (__bf16)0.0f : (__bf16)v;
        }
    }
}

// h = src1 @ W1 + shift(src2) @ W2 + b, written as f32.  16 WMMA per wave.
__device__ __forceinline__ void stage_sage(const __bf16* __restrict__ src1,
                                           const __bf16* __restrict__ src2,
                                           const __bf16* __restrict__ Wt1,
                                           const __bf16* __restrict__ Wt2,
                                           const float*  __restrict__ bias,
                                           float* __restrict__ dst,
                                           int wave, int n, int h) {
    int row  = wave * 16 + n;
    int row2 = row - 1; if (row2 < 0) row2 = 0;     // clamped row feeds an unused output
    const __bf16* ar1 = src1 + row  * BF_S;
    const __bf16* ar2 = src2 + row2 * BF_S;
    v16bf a0 = load_afrag(ar1, h, 0), a1 = load_afrag(ar1, h, 32);
    v16bf c0 = load_afrag(ar2, h, 0), c1 = load_afrag(ar2, h, 32);
#pragma unroll
    for (int nt = 0; nt < 4; ++nt) {
        const __bf16* w1 = Wt1 + (nt * 16 + n) * W_S;
        const __bf16* w2 = Wt2 + (nt * 16 + n) * W_S;
        float bv = bias[nt * 16 + n];
        v8f acc;
#pragma unroll
        for (int i = 0; i < 8; ++i) acc[i] = bv;
        acc = __builtin_amdgcn_wmma_f32_16x16x32_bf16(false, a0, false, load_bfrag(w1, h, 0),  (short)0, acc, false, false);
        acc = __builtin_amdgcn_wmma_f32_16x16x32_bf16(false, a1, false, load_bfrag(w1, h, 32), (short)0, acc, false, false);
        acc = __builtin_amdgcn_wmma_f32_16x16x32_bf16(false, c0, false, load_bfrag(w2, h, 0),  (short)0, acc, false, false);
        acc = __builtin_amdgcn_wmma_f32_16x16x32_bf16(false, c1, false, load_bfrag(w2, h, 32), (short)0, acc, false, false);
#pragma unroll
        for (int r = 0; r < 8; ++r)
            dst[(wave * 16 + r + 8 * h) * F_S + nt * 16 + n] = acc[r];
    }
}

__global__ __launch_bounds__(256, 1)
void fused_sage_chain(const float* __restrict__ loc, const float* __restrict__ dis,
                      const float* __restrict__ Wp1, const float* __restrict__ bp1,
                      const float* __restrict__ Ws1, const float* __restrict__ Wn1,
                      const float* __restrict__ bs1,
                      const float* __restrict__ Wp3, const float* __restrict__ bp3,
                      const float* __restrict__ Ws3, const float* __restrict__ Wn3,
                      const float* __restrict__ bs3,
                      float* __restrict__ out) {
    extern __shared__ char smem[];
    __bf16* sW[6];
    sW[0] = (__bf16*)smem;
    for (int w = 1; w < 6; ++w) sW[w] = sW[w - 1] + 64 * W_S;
    float*  sbias = (float*)(sW[5] + 64 * W_S);          // [bp1|bs1|bp3|bs3]
    __bf16* locb  = (__bf16*)(sbias + 256);
    __bf16* p1b   = locb + R_ROWS * BF_S;
    __bf16* s1b   = p1b  + R_ROWS * BF_S;
    __bf16* p3b   = s1b  + R_ROWS * BF_S;
    float*  h1f   = (float*)(p3b + R_ROWS * BF_S);
    float*  h2f   = h1f + R_ROWS * F_S;

    const int tid = threadIdx.x;
    const int b   = blockIdx.y;
    const int r0  = blockIdx.x * T_OUT;

    // Stage weights transposed+bf16 into LDS (L2-resident across all WGs).
    const float* wsrc[6] = {Wp1, Ws1, Wn1, Wp3, Ws3, Wn3};
#pragma unroll
    for (int w = 0; w < 6; ++w) {
        const float* Wg = wsrc[w];
        __bf16*      Wl = sW[w];
        for (int i = tid; i < 64 * 64; i += 256) {
            int d = i >> 6, f = i & 63;                 // W[d][f] -> Wt[f][d]
            Wl[f * W_S + d] = (__bf16)Wg[i];
        }
    }
    {
        const float* bsrc[4] = {bp1, bs1, bp3, bs3};
#pragma unroll
        for (int w = 0; w < 4; ++w)
            if (tid < 64) sbias[w * 64 + tid] = bsrc[w][tid];
    }
    // Stage loc tile (rows r0-2 .. r0+125) as bf16, zero-padded at bounds.
    for (int i = tid; i < R_ROWS * 64; i += 256) {
        int  lr = i >> 6, c = i & 63;
        long g  = (long)r0 - 2 + lr;
        float v = (g >= 0 && g < LSZ) ? loc[((size_t)b * LSZ + g) * 64 + c] : 0.0f;
        locb[lr * BF_S + c] = (__bf16)v;
    }
    __syncthreads();

    const int wave = tid >> 5, n = tid & 15, h = (tid >> 4) & 1;

    // Layer 1: p1 = relu(loc@Wp1+bp1); h1 = loc@Ws1 + shift(p1)@Wn1 + bs1
    stage_pool(locb, sW[0], sbias + 0, p1b, r0, wave, n, h);
    __syncthreads();
    stage_sage(locb, p1b, sW[1], sW[2], sbias + 64, h1f, wave, n, h);
    __syncthreads();

    // Edge product s1[l] = h1[l] * h1[l+1]  (zero where global index < 0)
    for (int i = tid; i < R_ROWS * 64; i += 256) {
        int  lr  = i >> 6, c = i & 63;
        int  lr2 = (lr == R_ROWS - 1) ? lr : lr + 1;   // row 127 unused downstream
        long g   = (long)r0 - 2 + lr;
        float v  = h1f[lr * F_S + c] * h1f[lr2 * F_S + c];
        s1b[lr * BF_S + c] = (g < 0) ? (__bf16)0.0f : (__bf16)v;
    }
    __syncthreads();

    // Layer 2 on s1
    stage_pool(s1b, sW[3], sbias + 128, p3b, r0, wave, n, h);
    __syncthreads();
    stage_sage(s1b, p3b, sW[4], sW[5], sbias + 192, h2f, wave, n, h);
    __syncthreads();

    // s2[l] = h2[l]*h2[l+1] -> out[b, l, 0:64]; local dist -> out[b, l, 64]
    for (int i = tid; i < T_OUT * 64; i += 256) {
        int j = i >> 6, c = i & 63;
        int l = r0 + j;
        if (l < LSZ - 2) {
            int lr = 2 + j;
            float v = h2f[lr * F_S + c] * h2f[(lr + 1) * F_S + c];
            out[((size_t)b * (LSZ - 2) + l) * 65 + c] = v;
        }
    }
    for (int j = tid; j < T_OUT; j += 256) {
        int l = r0 + j;
        if (l < LSZ - 2) {
            float d2 = (dis[(size_t)b * LSZ + l + 2] - dis[(size_t)b * LSZ + l]) * (1.0f / DIST_STD);
            out[((size_t)b * (LSZ - 2) + l) * 65 + 64] = (d2 - DIST_MEAN) * (1.0f / DIST_STD);
        }
    }
}

extern "C" void kernel_launch(void* const* d_in, const int* in_sizes, int n_in,
                              void* d_out, int out_size, void* d_ws, size_t ws_size,
                              hipStream_t stream) {
    (void)in_sizes; (void)n_in; (void)out_size; (void)d_ws; (void)ws_size;
    const float* loc = (const float*)d_in[0];
    const float* dis = (const float*)d_in[1];
    // d_in[2]=current_tim, d_in[3]=attr_t, d_in[4]=timeid: unused by reference output
    const float* Wp1 = (const float*)d_in[5];
    const float* bp1 = (const float*)d_in[6];
    const float* Ws1 = (const float*)d_in[7];
    const float* Wn1 = (const float*)d_in[8];
    const float* bs1 = (const float*)d_in[9];
    const float* Wp3 = (const float*)d_in[10];
    const float* bp3 = (const float*)d_in[11];
    const float* Ws3 = (const float*)d_in[12];
    const float* Wn3 = (const float*)d_in[13];
    const float* bs3 = (const float*)d_in[14];
    float* out = (float*)d_out;

    const int tiles = (LSZ - 2 + T_OUT - 1) / T_OUT;   // 34
    const size_t shmem =
        6 * 64 * W_S * sizeof(__bf16) +                // transposed weights
        256 * sizeof(float) +                          // biases
        4 * (size_t)R_ROWS * BF_S * sizeof(__bf16) +   // locb,p1b,s1b,p3b
        2 * (size_t)R_ROWS * F_S  * sizeof(float);     // h1f,h2f  (~209 KB)

    fused_sage_chain<<<dim3(tiles, BSZ), dim3(256), shmem, stream>>>(
        loc, dis, Wp1, bp1, Ws1, Wn1, bs1, Wp3, bp3, Ws3, Wn3, bs3, out);
}
